// DescriptorMatcher_82652350644933
// MI455X (gfx1250) — compile-verified, matrix operand order checked by reading, not run on hardware
//
#include <hip/hip_runtime.h>
#include <hip/hip_bf16.h>

// ---------------------------------------------------------------------------
// DescriptorMatcher (match_snn) for MI455X / gfx1250
//   dm = ||a||^2 + ||b||^2 - 2 a.b  ; cross term via bf16 WMMA, norms in fp32
//   fused per-row top-2 in GEMM epilogue; partial merge + ratio test pass
//
// GEMM kernel: 256x128 C-tile per block (8 waves). Each wave owns TWO 16-row
// A stripes sharing the same 8 B fragments -> 16 WMMAs per 10 fragment loads
// (2x the arithmetic intensity of the 1-stripe version; ~640 B WGP$ traffic
// per v_wmma instead of ~1150 B).
// ---------------------------------------------------------------------------

typedef __attribute__((ext_vector_type(16))) __bf16 v16bf;
typedef __attribute__((ext_vector_type(8)))  float  v8f;
typedef __attribute__((ext_vector_type(4)))  unsigned short us4;

#define NROWS 8192
#define NCOLS 8192
#define KDIM  512
#define COL_TILES (NCOLS / 128)   // 64 partials per row

// round-to-nearest-even fp32 -> bf16 (bit manipulation; no codegen risk)
static __device__ __forceinline__ unsigned short f32_to_bf16(float f) {
    unsigned int u = __float_as_uint(f);
    u += 0x7FFFu + ((u >> 16) & 1u);
    return (unsigned short)(u >> 16);
}

// ---- top-2 (smallest) helpers; only the nearest index is tracked ----------
static __device__ __forceinline__ void top2_insert(float& d0, int& i0, float& d1,
                                                   float v, int c) {
    if (v < d0) { d1 = d0; d0 = v; i0 = c; }
    else if (v < d1) { d1 = v; }
}

static __device__ __forceinline__ void top2_merge(float& d0, int& i0, float& d1,
                                                  float bd0, int bi0, float bd1) {
    if (bd0 < d0) {
        d1 = fminf(d0, bd1);
        d0 = bd0; i0 = bi0;
    } else {
        d1 = fminf(d1, bd0);
    }
}

// ---------------------------------------------------------------------------
// Kernel 1: fp32 -> bf16 conversion + fp32 row squared-norms.
// One wave per descriptor row; 8 waves / block; waves [0,8192) -> desc1.
// ---------------------------------------------------------------------------
__global__ __launch_bounds__(256) void dm_convert_norms(
    const float* __restrict__ desc1, const float* __restrict__ desc2,
    unsigned short* __restrict__ A, unsigned short* __restrict__ B,
    float* __restrict__ n1, float* __restrict__ n2)
{
    const int gw   = (int)((blockIdx.x * 256u + threadIdx.x) >> 5);
    const int lane = (int)(threadIdx.x & 31u);

    const float* src; unsigned short* dst; float* nrm; int row;
    if (gw < NROWS) { src = desc1; dst = A; nrm = n1; row = gw; }
    else            { src = desc2; dst = B; nrm = n2; row = gw - NROWS; }

    const float4* rp = (const float4*)(src + (size_t)row * KDIM);
    us4* wp = (us4*)(dst + (size_t)row * KDIM);

    float s = 0.0f;
    #pragma unroll
    for (int it = 0; it < 4; ++it) {
        const int e = it * 32 + lane;          // float4 index within row
        float4 x = rp[e];
        s += x.x * x.x + x.y * x.y + x.z * x.z + x.w * x.w;
        us4 o;
        o.x = f32_to_bf16(x.x);
        o.y = f32_to_bf16(x.y);
        o.z = f32_to_bf16(x.z);
        o.w = f32_to_bf16(x.w);
        wp[e] = o;
    }
    #pragma unroll
    for (int m = 1; m < 32; m <<= 1) s += __shfl_xor(s, m, 32);
    if (lane == 0) nrm[row] = s;
}

// ---------------------------------------------------------------------------
// Kernel 2: 256x128 C-tile bf16 WMMA GEMM with fused per-row top-2 epilogue.
// Block = 256 threads = 8 waves; wave w owns row stripes
//   [rowBase0, +16) and [rowBase0 + 128, +16)  with rowBase0 = 256*by + 16*w,
// and 8 WMMA column tiles (columns colBase + 16t). K loop: 16 x (K=32).
// A/B fragments: lane (l&15) = row/col, 16 contiguous bf16 K values
// (K-half selected by l>>4) -> plain 32B contiguous loads, no transpose.
// ---------------------------------------------------------------------------
__global__ __launch_bounds__(256) void dm_gemm_top2(
    const unsigned short* __restrict__ A, const unsigned short* __restrict__ B,
    const float* __restrict__ n1, const float* __restrict__ n2,
    float* __restrict__ p0, float* __restrict__ p1, int* __restrict__ pidx)
{
    const int lane = (int)(threadIdx.x & 31u);
    const int wave = (int)(threadIdx.x >> 5);
    const int l15  = lane & 15;
    const int half = lane >> 4;               // which K-half this lane loads

    const int colBase  = (int)blockIdx.x * 128;
    const int rowBase0 = (int)blockIdx.y * 256 + wave * 16;   // stripe 0
    // stripe 1 = rowBase0 + 128

    const unsigned short* aPtr0 = A + (size_t)(rowBase0 + l15) * KDIM + half * 16;
    const unsigned short* aPtr1 = aPtr0 + (size_t)128 * KDIM;
    const unsigned short* bPtr  = B + (size_t)(colBase + l15) * KDIM + half * 16;

    v8f acc0[8] = {};
    v8f acc1[8] = {};

    for (int kk = 0; kk < KDIM; kk += 32) {
        v16bf af0 = *(const v16bf*)(aPtr0 + kk);
        v16bf af1 = *(const v16bf*)(aPtr1 + kk);
        #pragma unroll
        for (int t = 0; t < 8; ++t) {
            v16bf bf = *(const v16bf*)(bPtr + (size_t)t * 16 * KDIM + kk);
            acc0[t] = __builtin_amdgcn_wmma_f32_16x16x32_bf16(
                false, af0, false, bf, (short)0, acc0[t], false, false);
            acc1[t] = __builtin_amdgcn_wmma_f32_16x16x32_bf16(
                false, af1, false, bf, (short)0, acc1[t], false, false);
        }
    }

    // ---- epilogue: squared distances + per-row top-2 over 128 columns ----
    float n2v[8];
    #pragma unroll
    for (int t = 0; t < 8; ++t) n2v[t] = n2[colBase + t * 16 + l15];

    const int colTile = (int)blockIdx.x;

    #pragma unroll
    for (int s = 0; s < 2; ++s) {
        const int rowBase = rowBase0 + s * 128;
        #pragma unroll
        for (int v = 0; v < 8; ++v) {
            // C layout: VGPR v = row v (lanes 0-15) and row v+8 (lanes 16-31)
            const int row = rowBase + v + (half << 3);
            const float n1v = n1[row];

            float d0 = 3.0e38f, d1 = 3.0e38f; int i0 = 0;
            #pragma unroll
            for (int t = 0; t < 8; ++t) {
                const float dot = s ? acc1[t][v] : acc0[t][v];
                float sq = fmaxf(n1v + n2v[t] - 2.0f * dot, 0.0f);
                top2_insert(d0, i0, d1, sq, colBase + t * 16 + l15);
            }
            // butterfly within each 16-lane half (one row lives in 16 lanes)
            #pragma unroll
            for (int m = 1; m < 16; m <<= 1) {
                float od0 = __shfl_xor(d0, m, 32);
                int   oi0 = __shfl_xor(i0, m, 32);
                float od1 = __shfl_xor(d1, m, 32);
                top2_merge(d0, i0, d1, od0, oi0, od1);
            }
            if (l15 == 0) {
                const size_t p = (size_t)row * COL_TILES + colTile;
                p0[p] = d0;
                p1[p] = d1;
                pidx[p] = i0;
            }
        }
    }
}

// ---------------------------------------------------------------------------
// Kernel 3: merge 64 partials per row, sqrt, Lowe ratio test, emit outputs.
// d_out layout (float-view): [0,8192) ratio-or-0 ; [8192,24576) int32 pairs
// (row, nn_idx) bit-cast ; [24576,32768) mask as 0/1.
// ---------------------------------------------------------------------------
__global__ __launch_bounds__(256) void dm_merge_finalize(
    const float* __restrict__ p0, const float* __restrict__ p1,
    const int* __restrict__ pidx, float* __restrict__ out)
{
    const int lane = (int)(threadIdx.x & 31u);
    const int wave = (int)(threadIdx.x >> 5);
    const int row  = (int)blockIdx.x * 8 + wave;

    const size_t base = (size_t)row * COL_TILES;
    float d0 = p0[base + lane];
    float d1 = p1[base + lane];
    int   i0 = pidx[base + lane];
    {
        float e0 = p0[base + 32 + lane];
        float e1 = p1[base + 32 + lane];
        int   j0 = pidx[base + 32 + lane];
        top2_merge(d0, i0, d1, e0, j0, e1);
    }
    #pragma unroll
    for (int m = 1; m < 32; m <<= 1) {
        float od0 = __shfl_xor(d0, m, 32);
        int   oi0 = __shfl_xor(i0, m, 32);
        float od1 = __shfl_xor(d1, m, 32);
        top2_merge(d0, i0, d1, od0, oi0, od1);
    }
    if (lane == 0) {
        const float a = sqrtf(d0);
        const float b = sqrtf(d1);
        const float ratio = a / b;
        const bool  ok = (ratio <= 0.8f);

        out[row] = ok ? ratio : 0.0f;                       // match_dists
        int* oi = (int*)(out + NROWS);                      // matches_idxs
        oi[2 * row]     = row;
        oi[2 * row + 1] = i0;
        float* om = out + NROWS + 2 * NROWS;                // mask
        om[row] = ok ? 1.0f : 0.0f;
    }
}

// ---------------------------------------------------------------------------
extern "C" void kernel_launch(void* const* d_in, const int* in_sizes, int n_in,
                              void* d_out, int out_size, void* d_ws, size_t ws_size,
                              hipStream_t stream) {
    const float* desc1 = (const float*)d_in[0];
    const float* desc2 = (const float*)d_in[1];
    float* out = (float*)d_out;

    // workspace carve-up (~22.3 MB total)
    char* ws = (char*)d_ws;
    unsigned short* A  = (unsigned short*)ws;                   //  8 MB bf16 desc1
    unsigned short* B  = A + (size_t)NROWS * KDIM;              //  8 MB bf16 desc2
    float* n1 = (float*)(B + (size_t)NCOLS * KDIM);             // 32 KB
    float* n2 = n1 + NROWS;                                     // 32 KB
    float* p0 = n2 + NCOLS;                                     //  2 MB sq0 partials
    float* p1 = p0 + (size_t)NROWS * COL_TILES;                 //  2 MB sq1 partials
    int*   pi = (int*)(p1 + (size_t)NROWS * COL_TILES);         //  2 MB idx partials

    // 1) convert + norms: 16384 waves -> 2048 blocks of 8 waves
    hipLaunchKernelGGL(dm_convert_norms, dim3(2048), dim3(256), 0, stream,
                       desc1, desc2, A, B, n1, n2);

    // 2) WMMA GEMM + fused top-2: 64x32 blocks of 256x128 C-tiles
    hipLaunchKernelGGL(dm_gemm_top2, dim3(NCOLS / 128, NROWS / 256), dim3(256), 0,
                       stream, A, B, n1, n2, p0, p1, pi);

    // 3) merge partials + ratio test: one wave per row
    hipLaunchKernelGGL(dm_merge_finalize, dim3(NROWS / 8), dim3(256), 0, stream,
                       p0, p1, pi, out);
}
